// DynamicGNN_11922829214219
// MI455X (gfx1250) — compile-verified, matrix operand order checked by reading, not run on hardware
//
#include <hip/hip_runtime.h>
#include <hip/hip_bf16.h>
#include <math.h>

// Problem constants (match reference)
#define Gn   64
#define Nn   1000
#define En   8192
#define Cc   128
#define Hh   4
#define Dd   32          // Cc / Hh
#define HIDn 256
#define EPSc 1e-5f

typedef __attribute__((ext_vector_type(16))) __bf16    bf16x16;
typedef __attribute__((ext_vector_type(8)))  float     f32x8;
typedef __attribute__((ext_vector_type(8)))  unsigned  u32x8;

__device__ __forceinline__ unsigned short f2bfu(float f) {
  unsigned u = __builtin_bit_cast(unsigned, f);
  unsigned r = u + 0x7FFFu + ((u >> 16) & 1u);   // round-to-nearest-even
  return (unsigned short)(r >> 16);
}
__device__ __forceinline__ __bf16 f2bf(float f) {
  return __builtin_bit_cast(__bf16, f2bfu(f));
}

// ---------------------------------------------------------------------------
// Degree / normalization
// ---------------------------------------------------------------------------
__global__ void k_init_deg(float* deg) {
  int i = blockIdx.x * blockDim.x + threadIdx.x;
  if (i < Gn * Nn) deg[i] = 1.0f;                // +1.0 self term folded in
}

__global__ void k_count_deg(const int* __restrict__ ei, float* deg) {
  int t = blockIdx.x * blockDim.x + threadIdx.x;
  if (t >= Gn * En) return;
  int g = t / En, e = t % En;
  int dst = ei[(size_t)g * 2 * En + En + e];
  atomicAdd(&deg[g * Nn + dst], 1.0f);
}

__global__ void k_dinv(const float* __restrict__ deg, float* dinv) {
  int i = blockIdx.x * blockDim.x + threadIdx.x;
  if (i < Gn * Nn) dinv[i] = rsqrtf(deg[i]);
}

// ---------------------------------------------------------------------------
// Pack W (128x128 f32, row-major K x N) into per-lane bf16 B fragments in the
// WMMA 16-bit B layout.  Fragment index t = (coltile*4 + kstep)*32 + lane,
// each fragment = 16 bf16 = 32 contiguous bytes.
// ---------------------------------------------------------------------------
__global__ void k_packB(const float* __restrict__ W, unsigned short* __restrict__ Bp) {
  int t = blockIdx.x * blockDim.x + threadIdx.x;     // 0..1023
  if (t >= 8 * 4 * 32) return;
  int lane = t & 31;
  int ks   = (t >> 5) & 3;
  int w    = t >> 7;
  int n  = w * 16 + (lane & 15);
  int ko = (lane >> 4) * 8;
  int kk = ks * 32;
  unsigned short* dst = Bp + (size_t)t * 16;
  #pragma unroll
  for (int j = 0; j < 8; ++j) {
    dst[j]     = f2bfu(W[(kk + ko + j) * Cc + n]);
    dst[j + 8] = f2bfu(W[(kk + 16 + ko + j) * Cc + n]);
  }
}

// ---------------------------------------------------------------------------
// WMMA GEMM:  out[g, m, c] = sum_k feat[g, m, k] * W[k, c]
// One wave computes one 16x16 tile (K=128 via 4x v_wmma_f32_16x16x32_bf16).
// Block = 256 threads = 8 waves -> 16 rows x 128 cols per block.
// grid = (63 row tiles, 64 graphs).  B comes pre-packed from k_packB.
// Tail rows are loaded from a clamped (valid) row unconditionally: WMMA row m
// of D depends only on row m of A, and the store guard drops rows >= Nn.
// ---------------------------------------------------------------------------
__global__ void __launch_bounds__(256)
k_gemm_wmma(const float* __restrict__ feat,
            const unsigned short* __restrict__ Bp,
            float* __restrict__ out) {
  const int mt   = blockIdx.x;
  const int g    = blockIdx.y;
  const int wave = threadIdx.x >> 5;   // col tile 0..7
  const int lane = threadIdx.x & 31;
  const int r    = lane & 15;
  const int hi   = lane >> 4;          // lane half
  const int ko   = hi * 8;
  const int colb = wave * 16;

  int m = mt * 16 + r;
  if (m >= Nn) m = Nn - 1;             // clamp: OOB-safe, rows discarded later

  const float4* arow4 = (const float4*)(feat + ((size_t)g * Nn + m) * Cc);
  const u32x8*  bp    = (const u32x8*)Bp;

  f32x8 acc = {};
  #pragma unroll
  for (int ks = 0; ks < 4; ++ks) {
    const int kk = ks * 32;
    // A fragment: two contiguous 8-float runs -> 4x global_load_b128
    float4 a0 = arow4[(kk + ko) >> 2];
    float4 a1 = arow4[((kk + ko) >> 2) + 1];
    float4 a2 = arow4[(kk + 16 + ko) >> 2];
    float4 a3 = arow4[((kk + 16 + ko) >> 2) + 1];
    bf16x16 a;
    a[0]  = f2bf(a0.x); a[1]  = f2bf(a0.y); a[2]  = f2bf(a0.z); a[3]  = f2bf(a0.w);
    a[4]  = f2bf(a1.x); a[5]  = f2bf(a1.y); a[6]  = f2bf(a1.z); a[7]  = f2bf(a1.w);
    a[8]  = f2bf(a2.x); a[9]  = f2bf(a2.y); a[10] = f2bf(a2.z); a[11] = f2bf(a2.w);
    a[12] = f2bf(a3.x); a[13] = f2bf(a3.y); a[14] = f2bf(a3.z); a[15] = f2bf(a3.w);
    // B fragment: one 32-byte coalesced load of pre-packed bf16
    bf16x16 b = __builtin_bit_cast(bf16x16, bp[(wave * 4 + ks) * 32 + lane]);
    acc = __builtin_amdgcn_wmma_f32_16x16x32_bf16(
        false, a, false, b, (short)0, acc, false, false);
  }

  #pragma unroll
  for (int v = 0; v < 8; ++v) {
    int mm = mt * 16 + v + 8 * hi;     // C/D layout: M = v + 8*(lane>>4)
    if (mm < Nn)
      out[((size_t)g * Nn + mm) * Cc + colb + (lane & 15)] = acc[v];
  }
}

// ---------------------------------------------------------------------------
// Zero a float buffer (vectorized)
// ---------------------------------------------------------------------------
__global__ void k_zero4(float4* p, int nvec) {
  int i = blockIdx.x * blockDim.x + threadIdx.x;
  if (i < nvec) p[i] = make_float4(0.f, 0.f, 0.f, 0.f);
}

// ---------------------------------------------------------------------------
// Edge scatter: aggr[g,dst,:] += hW[g,src,:] * dinv[src]*dinv[dst]
// One block per edge, one lane per feature channel.
// ---------------------------------------------------------------------------
__global__ void k_scatter(const float* __restrict__ hW,
                          const int* __restrict__ ei,
                          const float* __restrict__ dinv,
                          float* aggr) {
  int ge = blockIdx.x;
  int g = ge / En, e = ge % En;
  int src = ei[(size_t)g * 2 * En + e];
  int dst = ei[(size_t)g * 2 * En + En + e];
  float coef = dinv[g * Nn + src] * dinv[g * Nn + dst];
  int c = threadIdx.x;
  float v = hW[((size_t)g * Nn + src) * Cc + c] * coef;
  atomicAdd(&aggr[((size_t)g * Nn + dst) * Cc + c], v);
}

// ---------------------------------------------------------------------------
// feat_out = tanh(aggr + hW * dinv^2 + b)   (feat_out may alias aggr)
// ---------------------------------------------------------------------------
__global__ void k_finalize(const float* __restrict__ hW,
                           const float* __restrict__ aggr,
                           const float* __restrict__ dinv,
                           const float* __restrict__ bias,
                           float* feat_out) {
  int i = blockIdx.x * blockDim.x + threadIdx.x;
  if (i >= Gn * Nn * Cc) return;
  int c  = i % Cc;
  int gn = i / Cc;             // g*Nn + n
  float di = dinv[gn];
  feat_out[i] = tanhf(aggr[i] + hW[i] * di * di + bias[c]);
}

// ---------------------------------------------------------------------------
// Pool over nodes: gp[g,c] = sum_n feat[g,n,c]
// ---------------------------------------------------------------------------
__global__ void k_pool(const float* __restrict__ feat, float* gp) {
  int g = blockIdx.x, c = threadIdx.x;
  float s = 0.0f;
  for (int n = 0; n < Nn; ++n) s += feat[((size_t)g * Nn + n) * Cc + c];
  gp[g * Cc + c] = s;
}

// ---------------------------------------------------------------------------
// qkv[i,j] = in_b[j] + sum_k gp[i,k] * in_w[j,k]      (i<64, j<384)
// ---------------------------------------------------------------------------
__global__ void k_qkv(const float* __restrict__ gp,
                      const float* __restrict__ in_w,
                      const float* __restrict__ in_b,
                      float* qkv) {
  int t = blockIdx.x * blockDim.x + threadIdx.x;
  if (t >= Gn * 3 * Cc) return;
  int i = t / (3 * Cc), j = t % (3 * Cc);
  float s = in_b[j];
  const float* gr = gp + i * Cc;
  const float* wr = in_w + (size_t)j * Cc;
  for (int k = 0; k < Cc; ++k) s += gr[k] * wr[k];
  qkv[t] = s;
}

// ---------------------------------------------------------------------------
// Attention: one block per (q, head), 64 threads (one per key graph).
// o[q, h*32+d] = sum_k softmax_k(q.k/sqrt(d)) * v[k, h*32+d]
// ---------------------------------------------------------------------------
__global__ void k_attn(const float* __restrict__ qkv, float* o) {
  int b = blockIdx.x;
  int q = b / Hh, h = b % Hh;
  int t = threadIdx.x;          // 0..63
  __shared__ float sc[Gn];

  const float* qrow = qkv + q * (3 * Cc) + h * Dd;
  const float* krow = qkv + t * (3 * Cc) + Cc + h * Dd;
  float s = 0.0f;
  #pragma unroll
  for (int d = 0; d < Dd; ++d) s += qrow[d] * krow[d];
  sc[t] = s * 0.17677669529663687f;      // 1/sqrt(32)
  __syncthreads();

  float mx = -1e30f;
  for (int k = 0; k < Gn; ++k) mx = fmaxf(mx, sc[k]);
  float ex = __expf(sc[t] - mx);
  __syncthreads();
  sc[t] = ex;
  __syncthreads();
  float sum = 0.0f;
  for (int k = 0; k < Gn; ++k) sum += sc[k];

  if (t < Dd) {
    float acc = 0.0f;
    for (int k = 0; k < Gn; ++k)
      acc += sc[k] * qkv[k * (3 * Cc) + 2 * Cc + h * Dd + t];
    o[q * Cc + h * Dd + t] = acc / sum;
  }
}

// ---------------------------------------------------------------------------
// xatt[i,c] = out_b[c] + sum_k o[i,k] * out_w[c,k]
// ---------------------------------------------------------------------------
__global__ void k_xatt(const float* __restrict__ o,
                       const float* __restrict__ out_w,
                       const float* __restrict__ out_b,
                       float* xatt) {
  int t = blockIdx.x * blockDim.x + threadIdx.x;
  if (t >= Gn * Cc) return;
  int i = t / Cc, c = t % Cc;
  float s = out_b[c];
  const float* orow = o + i * Cc;
  const float* wr = out_w + (size_t)c * Cc;
  for (int k = 0; k < Cc; ++k) s += orow[k] * wr[k];
  xatt[t] = s;
}

// ---------------------------------------------------------------------------
// hid[i,j] = relu(mb1[j] + sum_k xatt[i,k] * mw1[k,j])
// ---------------------------------------------------------------------------
__global__ void k_hid(const float* __restrict__ xatt,
                      const float* __restrict__ mw1,
                      const float* __restrict__ mb1,
                      float* hid) {
  int t = blockIdx.x * blockDim.x + threadIdx.x;
  if (t >= Gn * HIDn) return;
  int i = t / HIDn, j = t % HIDn;
  float s = mb1[j];
  const float* xr = xatt + i * Cc;
  for (int k = 0; k < Cc; ++k) s += xr[k] * mw1[k * HIDn + j];
  hid[t] = fmaxf(s, 0.0f);
}

// ---------------------------------------------------------------------------
// Per graph: mlp2 + residual + layernorm + relu.  Block per graph, 128 thr.
// ---------------------------------------------------------------------------
__global__ void k_mlp_ln(const float* __restrict__ hid,
                         const float* __restrict__ mw2,
                         const float* __restrict__ mb2,
                         const float* __restrict__ xatt,
                         const float* __restrict__ ln_g,
                         const float* __restrict__ ln_b,
                         float* yr) {
  int g = blockIdx.x, c = threadIdx.x;
  __shared__ float buf[Cc];

  float s = mb2[c];
  const float* hr = hid + g * HIDn;
  for (int k = 0; k < HIDn; ++k) s += hr[k] * mw2[k * Cc + c];
  float pre = xatt[g * Cc + c] + s;
  buf[c] = pre;
  __syncthreads();

  float mu = 0.0f;
  for (int k = 0; k < Cc; ++k) mu += buf[k];
  mu *= (1.0f / Cc);
  float var = 0.0f;
  for (int k = 0; k < Cc; ++k) { float d = buf[k] - mu; var += d * d; }
  var *= (1.0f / Cc);

  float y = (pre - mu) * rsqrtf(var + EPSc) * ln_g[c] + ln_b[c];
  yr[g * Cc + c] = fmaxf(y, 0.0f);
}

// ---------------------------------------------------------------------------
// Final: node_repr[c] = sum_g yr[g,c];  out[j] = node_repr . lw[:,j] + lb[j]
// ---------------------------------------------------------------------------
__global__ void k_final(const float* __restrict__ yr,
                        const float* __restrict__ lw,
                        const float* __restrict__ lb,
                        float* out) {
  int c = threadIdx.x;                   // 0..127
  __shared__ float nr[Cc];
  float s = 0.0f;
  for (int g = 0; g < Gn; ++g) s += yr[g * Cc + c];
  nr[c] = s;
  __syncthreads();
  if (c < 2) {
    float acc = lb[c];
    for (int k = 0; k < Cc; ++k) acc += nr[k] * lw[k * 2 + c];
    out[c] = acc;
  }
}

// ---------------------------------------------------------------------------
extern "C" void kernel_launch(void* const* d_in, const int* in_sizes, int n_in,
                              void* d_out, int out_size, void* d_ws, size_t ws_size,
                              hipStream_t stream) {
  const float* x    = (const float*)d_in[0];
  const int*   ei   = (const int*)  d_in[1];
  const float* W0   = (const float*)d_in[2];
  const float* b0   = (const float*)d_in[3];
  const float* W1   = (const float*)d_in[4];
  const float* b1   = (const float*)d_in[5];
  const float* in_w = (const float*)d_in[6];
  const float* in_b = (const float*)d_in[7];
  const float* outw = (const float*)d_in[8];
  const float* outb = (const float*)d_in[9];
  const float* lng  = (const float*)d_in[10];
  const float* lnb  = (const float*)d_in[11];
  const float* mw1  = (const float*)d_in[12];
  const float* mb1  = (const float*)d_in[13];
  const float* mw2  = (const float*)d_in[14];
  const float* mb2  = (const float*)d_in[15];
  const float* lw   = (const float*)d_in[16];
  const float* lb   = (const float*)d_in[17];
  float* out = (float*)d_out;

  // Workspace layout (floats)
  float* ws   = (float*)d_ws;
  const size_t BIG = (size_t)Gn * Nn * Cc;        // 8,192,000
  float* buf1 = ws;                               // hW
  float* buf2 = buf1 + BIG;                       // feat / aggr (in place)
  float* deg  = buf2 + BIG;                       // 64,000
  float* dinv = deg  + (size_t)Gn * Nn;           // 64,000
  float* gp   = dinv + (size_t)Gn * Nn;           // 8,192
  float* qkv  = gp   + (size_t)Gn * Cc;           // 24,576
  float* obuf = qkv  + (size_t)Gn * 3 * Cc;       // 8,192
  float* xatt = obuf + (size_t)Gn * Cc;           // 8,192
  float* hid  = xatt + (size_t)Gn * Cc;           // 16,384
  float* yr   = hid  + (size_t)Gn * HIDn;         // 8,192
  unsigned short* Bp = (unsigned short*)(yr + (size_t)Gn * Cc); // 16,384 bf16

  const int GN  = Gn * Nn;
  const int GE  = Gn * En;
  const int GNC = (int)BIG;

  // Degree + normalization (shared by both layers)
  k_init_deg <<<(GN + 255) / 256, 256, 0, stream>>>(deg);
  k_count_deg<<<(GE + 255) / 256, 256, 0, stream>>>(ei, deg);
  k_dinv     <<<(GN + 255) / 256, 256, 0, stream>>>(deg, dinv);

  dim3 ggrid((Nn + 15) / 16, Gn);   // 63 x 64

  // ---- GCN layer 0 ----
  k_packB    <<<4, 256, 0, stream>>>(W0, Bp);
  k_gemm_wmma<<<ggrid, 256, 0, stream>>>(x, Bp, buf1);
  k_zero4    <<<(GNC / 4 + 255) / 256, 256, 0, stream>>>((float4*)buf2, GNC / 4);
  k_scatter  <<<GE, Cc, 0, stream>>>(buf1, ei, dinv, buf2);
  k_finalize <<<(GNC + 255) / 256, 256, 0, stream>>>(buf1, buf2, dinv, b0, buf2);

  // ---- GCN layer 1 ----
  k_packB    <<<4, 256, 0, stream>>>(W1, Bp);
  k_gemm_wmma<<<ggrid, 256, 0, stream>>>(buf2, Bp, buf1);
  k_zero4    <<<(GNC / 4 + 255) / 256, 256, 0, stream>>>((float4*)buf2, GNC / 4);
  k_scatter  <<<GE, Cc, 0, stream>>>(buf1, ei, dinv, buf2);
  k_finalize <<<(GNC + 255) / 256, 256, 0, stream>>>(buf1, buf2, dinv, b1, buf2);

  // ---- Head ----
  k_pool  <<<Gn, Cc, 0, stream>>>(buf2, gp);
  k_qkv   <<<(Gn * 3 * Cc + 255) / 256, 256, 0, stream>>>(gp, in_w, in_b, qkv);
  k_attn  <<<Gn * Hh, Gn, 0, stream>>>(qkv, obuf);
  k_xatt  <<<(Gn * Cc + 255) / 256, 256, 0, stream>>>(obuf, outw, outb, xatt);
  k_hid   <<<(Gn * HIDn + 255) / 256, 256, 0, stream>>>(xatt, mw1, mb1, hid);
  k_mlp_ln<<<Gn, Cc, 0, stream>>>(hid, mw2, mb2, xatt, lng, lnb, yr);
  k_final <<<1, Cc, 0, stream>>>(yr, lw, lb, out);
}
